// HeteroGNN_75582834475222
// MI455X (gfx1250) — matrix operand order repaired and checked by reading, not compiled, not run on hardware
//
#include <hip/hip_runtime.h>
#include <hip/hip_bf16.h>
#include <stdint.h>

// ---------------------------------------------------------------------------
// Problem constants (match reference)
// ---------------------------------------------------------------------------
static constexpr int kNG  = 50000;
static constexpr int kND  = 10000;
static constexpr int kIN  = 128;
static constexpr int kHID = 256;

typedef __bf16 bf16_t;
typedef __attribute__((ext_vector_type(16))) __bf16 v16bf;
typedef __attribute__((ext_vector_type(8)))  float  v8f;

union Frag16 { v16bf v; uint4 u[2]; };
static_assert(sizeof(v16bf) == 32, "v16bf must be 32 bytes");

// ---------------------------------------------------------------------------
// Edge scatter: one wave32 per edge. float4 gather + scalar f32 atomics.
// Degree accumulated by lane 0. d is a multiple of 128.
// ---------------------------------------------------------------------------
__global__ __launch_bounds__(256)
void scatter_edges(const float* __restrict__ H, const int* __restrict__ src,
                   const int* __restrict__ dst, int nE, int d,
                   float* __restrict__ agg, float* __restrict__ deg)
{
    int e = blockIdx.x * 8 + (threadIdx.x >> 5);
    if (e >= nE) return;
    int lane = threadIdx.x & 31;
    int s = src[e], t = dst[e];
    if (lane == 0) atomicAdd(deg + t, 1.0f);
    const float4* hs = reinterpret_cast<const float4*>(H + (size_t)s * d);
    float* ag = agg + (size_t)t * d;
    for (int k4 = lane; k4 * 4 < d; k4 += 32) {
        float4 v = hs[k4];
        int k = k4 * 4;
        atomicAdd(ag + k + 0, v.x);
        atomicAdd(ag + k + 1, v.y);
        atomicAdd(ag + k + 2, v.z);
        atomicAdd(ag + k + 3, v.w);
    }
}

// ---------------------------------------------------------------------------
// Pack fp32 activations -> bf16 A-panel at column offset; optional mean by deg.
// ---------------------------------------------------------------------------
__global__ __launch_bounds__(256)
void pack_act(const float* __restrict__ X, const float* __restrict__ deg,
              int M, int d, bf16_t* __restrict__ dst, int ldDst, int colOff)
{
    size_t total = (size_t)M * d;
    for (size_t i = (size_t)blockIdx.x * 256 + threadIdx.x; i < total;
         i += (size_t)gridDim.x * 256) {
        int r = (int)(i / (unsigned)d);
        int c = (int)(i % (unsigned)d);
        float v = X[i];
        if (deg) v *= 1.0f / fmaxf(deg[r], 1.0f);
        dst[(size_t)r * ldDst + colOff + c] = (bf16_t)v;
    }
}

// ---------------------------------------------------------------------------
// Pack weight (Kd x N fp32, row-major) transposed into Wt (N x Ktot bf16)
// at row offset rowOff; optional second weight added (fused self-weights).
// ---------------------------------------------------------------------------
__global__ __launch_bounds__(256)
void pack_wt(const float* __restrict__ W, const float* __restrict__ W2,
             int Kd, int N, bf16_t* __restrict__ Wt, int Ktot, int rowOff)
{
    size_t total = (size_t)Kd * N;
    for (size_t i = (size_t)blockIdx.x * 256 + threadIdx.x; i < total;
         i += (size_t)gridDim.x * 256) {
        int k = (int)(i / (unsigned)N);
        int n = (int)(i % (unsigned)N);
        float v = W[i];
        if (W2) v += W2[i];
        Wt[(size_t)n * Ktot + rowOff + k] = (bf16_t)v;
    }
}

__global__ __launch_bounds__(256)
void combine_bias(const float* __restrict__ b1, const float* __restrict__ b2,
                  int n, float* __restrict__ out)
{
    int i = threadIdx.x;
    if (i < n) out[i] = b1[i] + (b2 ? b2[i] : 0.0f);
}

// ---------------------------------------------------------------------------
// bf16 WMMA GEMM: C[M,N] = A[M,K] @ Wt[N,K]^T + bias, fp32 accumulate.
// Wave tile = 16 rows x 64 cols (4x v_wmma_f32_16x16x32_bf16 accumulators).
// Block = 256 threads = 8 waves covering 128 rows. grid = (ceil(M/128), N/64).
//
// A fragment: lane holds row lane%16, contiguous 16 K's at laneHalf*16
// -> base += laneHalf*16, chunks at +0 and +8.
// B fragment: lane holds column lane%16, contiguous 16 K's at laneHalf*16.
// ---------------------------------------------------------------------------
__global__ __launch_bounds__(256)
void gemm_bf16_wmma(const bf16_t* __restrict__ A,   // M x K row-major
                    const bf16_t* __restrict__ Wt,  // N x K row-major
                    const float*  __restrict__ bias,// N
                    float* __restrict__ C,          // M x N row-major
                    int M, int N, int K)
{
    const int tid      = threadIdx.x;
    const int wave     = tid >> 5;
    const int lane     = tid & 31;
    const int lane16   = lane & 15;
    const int laneHalf = lane >> 4;

    const int row0 = (blockIdx.x * 8 + wave) * 16;
    if (row0 >= M) return;                 // uniform per wave
    const int col0 = blockIdx.y * 64;

    int arow = row0 + lane16;
    if (arow >= M) arow = M - 1;           // clamp: keep EXEC all-ones for WMMA
    const bf16_t* aptr  = A + (size_t)arow * K + laneHalf * 16;
    const bf16_t* bptr0 = Wt + (size_t)(col0 +  0 + lane16) * K + laneHalf * 16;
    const bf16_t* bptr1 = Wt + (size_t)(col0 + 16 + lane16) * K + laneHalf * 16;
    const bf16_t* bptr2 = Wt + (size_t)(col0 + 32 + lane16) * K + laneHalf * 16;
    const bf16_t* bptr3 = Wt + (size_t)(col0 + 48 + lane16) * K + laneHalf * 16;

    v8f acc0 = {}, acc1 = {}, acc2 = {}, acc3 = {};

    for (int k0 = 0; k0 < K; k0 += 32) {
        Frag16 a, b0, b1, b2, b3;
        a.u[0]  = *reinterpret_cast<const uint4*>(aptr  + k0);
        a.u[1]  = *reinterpret_cast<const uint4*>(aptr  + k0 + 8);
        b0.u[0] = *reinterpret_cast<const uint4*>(bptr0 + k0);
        b0.u[1] = *reinterpret_cast<const uint4*>(bptr0 + k0 + 8);
        b1.u[0] = *reinterpret_cast<const uint4*>(bptr1 + k0);
        b1.u[1] = *reinterpret_cast<const uint4*>(bptr1 + k0 + 8);
        b2.u[0] = *reinterpret_cast<const uint4*>(bptr2 + k0);
        b2.u[1] = *reinterpret_cast<const uint4*>(bptr2 + k0 + 8);
        b3.u[0] = *reinterpret_cast<const uint4*>(bptr3 + k0);
        b3.u[1] = *reinterpret_cast<const uint4*>(bptr3 + k0 + 8);

        acc0 = __builtin_amdgcn_wmma_f32_16x16x32_bf16(false, a.v, false, b0.v,
                                                       (short)0, acc0, false, false);
        acc1 = __builtin_amdgcn_wmma_f32_16x16x32_bf16(false, a.v, false, b1.v,
                                                       (short)0, acc1, false, false);
        acc2 = __builtin_amdgcn_wmma_f32_16x16x32_bf16(false, a.v, false, b2.v,
                                                       (short)0, acc2, false, false);
        acc3 = __builtin_amdgcn_wmma_f32_16x16x32_bf16(false, a.v, false, b3.v,
                                                       (short)0, acc3, false, false);
    }

    // C/D layout: VGPR r -> M = row0 + r + 8*laneHalf, N = col tile + lane16.
    const int mbase = row0 + laneHalf * 8;
    const int cb = col0 + lane16;
#pragma unroll
    for (int r = 0; r < 8; ++r) {
        int row = mbase + r;
        if (row < M) {
            size_t base = (size_t)row * N + cb;
            C[base +  0] = acc0[r] + bias[cb +  0];
            C[base + 16] = acc1[r] + bias[cb + 16];
            C[base + 32] = acc2[r] + bias[cb + 32];
            C[base + 48] = acc3[r] + bias[cb + 48];
        }
    }
}

// ---------------------------------------------------------------------------
// BatchNorm column stats (biased var) -> per-column scale/shift.
// One block (256 threads) per column; LDS tree reduction.
// ---------------------------------------------------------------------------
__global__ __launch_bounds__(256)
void bn_stats(const float* __restrict__ X, int M, int N,
              const float* __restrict__ gamma, const float* __restrict__ beta,
              float* __restrict__ scale, float* __restrict__ shift)
{
    __shared__ float ss[256];
    __shared__ float sq[256];
    int col = blockIdx.x;
    float s = 0.f, q = 0.f;
    for (int r = threadIdx.x; r < M; r += 256) {
        float v = X[(size_t)r * N + col];
        s += v; q += v * v;
    }
    ss[threadIdx.x] = s; sq[threadIdx.x] = q;
    __syncthreads();
    for (int off = 128; off > 0; off >>= 1) {
        if ((int)threadIdx.x < off) {
            ss[threadIdx.x] += ss[threadIdx.x + off];
            sq[threadIdx.x] += sq[threadIdx.x + off];
        }
        __syncthreads();
    }
    if (threadIdx.x == 0) {
        float mean = ss[0] / (float)M;
        float var  = sq[0] / (float)M - mean * mean;
        float rstd = rsqrtf(var + 1e-5f);
        float sc   = gamma[col] * rstd;
        scale[col] = sc;
        shift[col] = beta[col] - mean * sc;
    }
}

__global__ __launch_bounds__(256)
void bn_relu(const float* __restrict__ X, const float* __restrict__ scale,
             const float* __restrict__ shift, int M, int N, float* __restrict__ H)
{
    size_t total = (size_t)M * N;
    for (size_t i = (size_t)blockIdx.x * 256 + threadIdx.x; i < total;
         i += (size_t)gridDim.x * 256) {
        int c = (int)(i % (unsigned)N);
        H[i] = fmaxf(X[i] * scale[c] + shift[c], 0.0f);
    }
}

// ---------------------------------------------------------------------------
// Host orchestration
// ---------------------------------------------------------------------------
extern "C" void kernel_launch(void* const* d_in, const int* in_sizes, int n_in,
                              void* d_out, int out_size, void* d_ws, size_t ws_size,
                              hipStream_t stream)
{
    (void)n_in; (void)out_size; (void)ws_size;

    const float* x_gene = (const float*)d_in[0];
    const float* x_dis  = (const float*)d_in[1];
    const int* src_gg = (const int*)d_in[2];
    const int* dst_gg = (const int*)d_in[3];
    const int* src_gd = (const int*)d_in[4];
    const int* dst_gd = (const int*)d_in[5];
    const int* src_dg = (const int*)d_in[6];
    const int* dst_dg = (const int*)d_in[7];
    const int E_gg = in_sizes[2], E_gd = in_sizes[4], E_dg = in_sizes[6];

    // params flattened in insertion order; layer base = 8 + 13*l:
    //  +0 Wself_gg +1 Wneigh_gg +2 b_gg | +3 Wself_gd +4 Wneigh_gd +5 b_gd
    //  +6 Wself_dg +7 Wneigh_dg +8 b_dg | +9 gamma_g +10 beta_g +11 gamma_d +12 beta_d
    // then 34 Wp_gene, 35 bp_gene, 36 Wp_disease, 37 bp_disease
    auto P = [&](int idx) { return (const float*)d_in[idx]; };

    // Workspace bump allocator (256B aligned)
    char* wp = (char*)d_ws;
    auto alloc = [&](size_t bytes) -> void* {
        void* r = (void*)wp;
        wp += (bytes + 255) & ~(size_t)255;
        return r;
    };
    float*  hg      = (float*) alloc((size_t)kNG * kHID * 4);
    float*  hd      = (float*) alloc((size_t)kND * kHID * 4);
    float*  out_g   = (float*) alloc((size_t)kNG * kHID * 4);
    float*  out_d   = (float*) alloc((size_t)kND * kHID * 4);
    float*  agg     = (float*) alloc((size_t)kNG * kHID * 4);
    float*  deg     = (float*) alloc((size_t)kNG * 4);
    bf16_t* packA_g = (bf16_t*)alloc((size_t)kNG * 3 * kHID * 2);
    bf16_t* packA_d = (bf16_t*)alloc((size_t)kND * 2 * kHID * 2);
    bf16_t* packW_g = (bf16_t*)alloc((size_t)kHID * 3 * kHID * 2);
    bf16_t* packW_d = (bf16_t*)alloc((size_t)kHID * 2 * kHID * 2);
    bf16_t* packWp  = (bf16_t*)alloc((size_t)kIN * kHID * 2);
    float*  bias_g  = (float*) alloc(kHID * 4);
    float*  bias_d  = (float*) alloc(kHID * 4);
    float*  scale   = (float*) alloc(kHID * 4);
    float*  shift   = (float*) alloc(kHID * 4);

    const int PKB = 4096;   // grid-stride blocks for pack/elementwise kernels

    for (int l = 0; l < 2; ++l) {
        const float* in_g = (l == 0) ? x_gene : hg;
        const float* in_d = (l == 0) ? x_dis  : hd;
        const int d = (l == 0) ? kIN : kHID;
        const int Ktot_g = 3 * d, Ktot_d = 2 * d;
        const int base = 8 + 13 * l;

        // ---- gene <- gene (mean over gg) -> packA_g cols [d, 2d)
        hipMemsetAsync(agg, 0, (size_t)kNG * d * 4, stream);
        hipMemsetAsync(deg, 0, (size_t)kNG * 4, stream);
        scatter_edges<<<(E_gg + 7) / 8, 256, 0, stream>>>(in_g, src_gg, dst_gg, E_gg, d, agg, deg);
        pack_act<<<PKB, 256, 0, stream>>>(agg, deg, kNG, d, packA_g, Ktot_g, d);

        // ---- gene <- disease (mean over dg) -> packA_g cols [2d, 3d)
        hipMemsetAsync(agg, 0, (size_t)kNG * d * 4, stream);
        hipMemsetAsync(deg, 0, (size_t)kNG * 4, stream);
        scatter_edges<<<(E_dg + 7) / 8, 256, 0, stream>>>(in_d, src_dg, dst_dg, E_dg, d, agg, deg);
        pack_act<<<PKB, 256, 0, stream>>>(agg, deg, kNG, d, packA_g, Ktot_g, 2 * d);

        // ---- gene self -> packA_g cols [0, d)
        pack_act<<<PKB, 256, 0, stream>>>(in_g, nullptr, kNG, d, packA_g, Ktot_g, 0);

        // ---- disease <- gene (mean over gd) -> packA_d cols [d, 2d); self [0, d)
        hipMemsetAsync(agg, 0, (size_t)kND * d * 4, stream);
        hipMemsetAsync(deg, 0, (size_t)kND * 4, stream);
        scatter_edges<<<(E_gd + 7) / 8, 256, 0, stream>>>(in_g, src_gd, dst_gd, E_gd, d, agg, deg);
        pack_act<<<PKB, 256, 0, stream>>>(agg, deg, kND, d, packA_d, Ktot_d, d);
        pack_act<<<PKB, 256, 0, stream>>>(in_d, nullptr, kND, d, packA_d, Ktot_d, 0);

        // ---- weights (transposed bf16), self-weights for gene fused (gg+dg)
        pack_wt<<<PKB, 256, 0, stream>>>(P(base + 0), P(base + 6), d, kHID, packW_g, Ktot_g, 0);
        pack_wt<<<PKB, 256, 0, stream>>>(P(base + 1), nullptr,     d, kHID, packW_g, Ktot_g, d);
        pack_wt<<<PKB, 256, 0, stream>>>(P(base + 7), nullptr,     d, kHID, packW_g, Ktot_g, 2 * d);
        pack_wt<<<PKB, 256, 0, stream>>>(P(base + 3), nullptr,     d, kHID, packW_d, Ktot_d, 0);
        pack_wt<<<PKB, 256, 0, stream>>>(P(base + 4), nullptr,     d, kHID, packW_d, Ktot_d, d);
        combine_bias<<<1, 256, 0, stream>>>(P(base + 2), P(base + 8), kHID, bias_g);
        combine_bias<<<1, 256, 0, stream>>>(P(base + 5), nullptr,     kHID, bias_d);

        // ---- GEMMs (WMMA bf16, fp32 accumulate)
        dim3 gridG((kNG + 127) / 128, kHID / 64);
        gemm_bf16_wmma<<<gridG, 256, 0, stream>>>(packA_g, packW_g, bias_g, out_g, kNG, kHID, Ktot_g);
        dim3 gridD((kND + 127) / 128, kHID / 64);
        gemm_bf16_wmma<<<gridD, 256, 0, stream>>>(packA_d, packW_d, bias_d, out_d, kND, kHID, Ktot_d);

        // ---- BN (batch stats) + ReLU
        bn_stats<<<kHID, 256, 0, stream>>>(out_g, kNG, kHID, P(base + 9), P(base + 10), scale, shift);
        bn_relu<<<PKB, 256, 0, stream>>>(out_g, scale, shift, kNG, kHID, hg);
        bn_stats<<<kHID, 256, 0, stream>>>(out_d, kND, kHID, P(base + 11), P(base + 12), scale, shift);
        bn_relu<<<PKB, 256, 0, stream>>>(out_d, scale, shift, kND, kHID, hd);
    }

    // ---- Final projections -> d_out = [og (NG x IN) | od (ND x IN)]
    float* og = (float*)d_out;
    float* od = og + (size_t)kNG * kIN;

    pack_act<<<PKB, 256, 0, stream>>>(hg, nullptr, kNG, kHID, packA_g, kHID, 0);
    pack_wt<<<PKB, 256, 0, stream>>>(P(34), nullptr, kHID, kIN, packWp, kHID, 0);
    dim3 gridPg((kNG + 127) / 128, kIN / 64);
    gemm_bf16_wmma<<<gridPg, 256, 0, stream>>>(packA_g, packWp, P(35), og, kNG, kIN, kHID);

    pack_act<<<PKB, 256, 0, stream>>>(hd, nullptr, kND, kHID, packA_d, kHID, 0);
    pack_wt<<<PKB, 256, 0, stream>>>(P(36), nullptr, kHID, kIN, packWp, kHID, 0);
    dim3 gridPd((kND + 127) / 128, kIN / 64);
    gemm_bf16_wmma<<<gridPd, 256, 0, stream>>>(packA_d, packWp, P(37), od, kND, kIN, kHID);
}
